// MambaFormerBlock_19885698581122
// MI455X (gfx1250) — compile-verified, matrix operand order checked by reading, not run on hardware
//
#include <hip/hip_runtime.h>
#include <hip/hip_bf16.h>

// ---------------------------------------------------------------------------
// MambaFormer block forward for gfx1250 (MI455X).
// Big GEMMs: v_wmma_f32_16x16x32_bf16, B tiles staged to LDS via the Tensor
// Data Mover (tensor_load_to_lds, TENSORcnt) with double buffering.
// ---------------------------------------------------------------------------

typedef __attribute__((ext_vector_type(16))) __bf16 v16bf;
typedef __attribute__((ext_vector_type(8)))  float  v8f;
typedef __attribute__((ext_vector_type(4)))  unsigned int v4u;
typedef __attribute__((ext_vector_type(4)))  int v4i;
typedef __attribute__((ext_vector_type(8)))  int v8i;

#define EPI_BIAS  1
#define EPI_RELU  2
#define EPI_SIG   4
#define EPI_RES   8
#define EPI_GATE  16
#define EPI_WF32  32
#define EPI_WBF16 64

#if defined(__has_builtin)
#if __has_builtin(__builtin_amdgcn_tensor_load_to_lds) && __has_builtin(__builtin_amdgcn_s_wait_tensorcnt)
#define HAVE_TDM 1
#endif
#endif
#ifndef HAVE_TDM
#define HAVE_TDM 0
#endif

__device__ __forceinline__ __bf16 f2bf(float f) {
    unsigned u = __float_as_uint(f);
    unsigned r = (u + 0x7FFFu + ((u >> 16) & 1u)) >> 16;   // RNE
    unsigned short h = (unsigned short)r;
    __bf16 out;
    __builtin_memcpy(&out, &h, 2);
    return out;
}

__device__ __forceinline__ float waveReduceSum(float v) {
    for (int o = 16; o > 0; o >>= 1) v += __shfl_down(v, o, 32);
    return v;
}
__device__ __forceinline__ float waveReduceMax(float v) {
    for (int o = 16; o > 0; o >>= 1) v = fmaxf(v, __shfl_down(v, o, 32));
    return v;
}

__device__ float blockReduceSum(float v) {
    __shared__ float sm[32];
    int lane = threadIdx.x & 31, wid = threadIdx.x >> 5;
    v = waveReduceSum(v);
    __syncthreads();
    if (lane == 0) sm[wid] = v;
    __syncthreads();
    int nw = (blockDim.x + 31) >> 5;
    float r = (threadIdx.x < (unsigned)nw) ? sm[threadIdx.x] : 0.f;
    if (wid == 0) r = waveReduceSum(r);
    if (threadIdx.x == 0) sm[0] = r;
    __syncthreads();
    return sm[0];
}

__device__ float blockReduceMax(float v) {
    __shared__ float sm[32];
    int lane = threadIdx.x & 31, wid = threadIdx.x >> 5;
    v = waveReduceMax(v);
    __syncthreads();
    if (lane == 0) sm[wid] = v;
    __syncthreads();
    int nw = (blockDim.x + 31) >> 5;
    float r = (threadIdx.x < (unsigned)nw) ? sm[threadIdx.x] : -3.0e38f;
    if (wid == 0) r = waveReduceMax(r);
    if (threadIdx.x == 0) sm[0] = r;
    __syncthreads();
    return sm[0];
}

// -------------------- WMMA fragment load ------------------------------------
// 16-bit A-matrix 16x32 layout (ISA 7.12.2): lanes 0-15 hold K 0..7 (v0..3)
// and K 16..23 (v4..7); lanes 16-31 hold K 8..15 and 24..31. B is stored
// column-major (N x K) so its fragment loads mirror A's. Works for both
// global (global_load_b128) and LDS (ds_load_b128) pointers.
__device__ __forceinline__ v16bf load_frag(const __bf16* row, int koff) {
    union { uint4 u[2]; v16bf v; } f;
    f.u[0] = *reinterpret_cast<const uint4*>(row + koff);
    f.u[1] = *reinterpret_cast<const uint4*>(row + koff + 16);
    return f.v;
}

#if HAVE_TDM
// Issue one TDM transfer: 64 rows x 32 bf16 (64B/row) tile of Bt -> LDS.
// D# bitfields per CDNA5 ISA ch.8: group0 {count=1, lds_addr, global_addr,
// type=2}; group1 {data_size=1(2B), tensor_dim0/1, tile_dim0=32, tile_dim1=64,
// tensor_dim0_stride=ldb}. Groups 2/3 (and the extra operand of the 6-arg
// builtin) are zero: 2-D tensor, no gather/iterate.
__device__ __forceinline__ void tdm_load_tile(unsigned lds_off, const __bf16* gaddr,
                                              int rows, int ld_elems) {
    unsigned long long ga = (unsigned long long)gaddr;
    v4u g0;
    g0.x = 1u;                                           // count=1, user mode
    g0.y = lds_off;                                      // lds_addr (bytes)
    g0.z = (unsigned)(ga & 0xFFFFFFFFu);                 // global_addr[31:0]
    g0.w = (unsigned)((ga >> 32) & 0x01FFFFFFu) | (2u << 30);  // addr[56:32] | type=2
    unsigned td0 = (unsigned)ld_elems;                   // tensor_dim0 >= tile
    unsigned td1 = (unsigned)rows;                       // tensor_dim1
    unsigned long long st0 = (unsigned long long)ld_elems;     // dim0 stride
    v8i g1;
    g1[0] = (int)(1u << 16);                             // data_size=1 (2 bytes)
    g1[1] = (int)((td0 & 0xFFFFu) << 16);                // tensor_dim0[15:0]
    g1[2] = (int)((td0 >> 16) | ((td1 & 0xFFFFu) << 16));// dim0 hi | dim1 lo
    g1[3] = (int)((td1 >> 16) | (32u << 16));            // dim1 hi | tile_dim0=32
    g1[4] = (int)(rows & 0xFFFF);                        // tile_dim1, tile_dim2=0
    g1[5] = (int)(st0 & 0xFFFFFFFFu);                    // dim0_stride[31:0]
    g1[6] = (int)((st0 >> 32) & 0xFFFFu);                // dim0_stride[47:32]
    g1[7] = 0;                                           // dim1_stride (unused, 2D)
    v4i z4 = {0, 0, 0, 0};
    v8i z8 = {0, 0, 0, 0, 0, 0, 0, 0};
    __builtin_amdgcn_tensor_load_to_lds(g0, g1, z4, z4, z8, 0);
}
#endif

// -------------------- Generic batched bf16 WMMA GEMM ------------------------
// C[M,N] = A[M,K] @ Bt[N,K]^T  (+ bias/gate/res epilogue).
// Block = 4 waves stacked along M; all waves share the 64-col B tile, which is
// staged in LDS by the TDM (double buffered). M, N multiples of 64; K of 32.
template <int EPI>
__global__ __launch_bounds__(128) void gemm_k(
    const __bf16* __restrict__ A, const __bf16* __restrict__ Bt,
    const float* __restrict__ bias, const float* __restrict__ gatep,
    const float* __restrict__ resp, int ldaux,
    float* __restrict__ Cf, int ldcf, __bf16* __restrict__ Cbf, int ldcbf,
    int M, int N, int K, int lda, int ldb,
    int H, long sAb, long sAh, long sBb, long sBh, long sCb, long sCh)
{
    __shared__ __align__(16) unsigned char bstage[2][4096];  // 2 x (64 rows x 64B)

    int z  = blockIdx.z;
    int hh = (H > 1) ? (z % H) : 0;
    long bb = (H > 1) ? (z / H) : (long)z;
    const __bf16* Ab = A + bb * sAb + (long)hh * sAh;
    const __bf16* Bb = Bt + bb * sBb + (long)hh * sBh;
    long coff = bb * sCb + (long)hh * sCh;

    int lane = threadIdx.x & 31;
    int wid  = threadIdx.x >> 5;
    int tm = blockIdx.x * 64 + wid * 16;
    int tn = blockIdx.y * 64;

    int ml = lane & 15;
    int kb = (lane >> 4) * 8;

    const __bf16* arow   = Ab + (size_t)(tm + ml) * lda;
    const __bf16* btile0 = Bb + (size_t)tn * ldb;   // rows tn..tn+63 of Bt
    int nk = K >> 5;

    v8f acc[4] = {};

#if HAVE_TDM
    if (wid == 0)
        tdm_load_tile((unsigned)(unsigned long long)&bstage[0][0], btile0, 64, ldb);
#endif

    for (int kt = 0; kt < nk; ++kt) {
        int k0 = kt << 5;
        int cur = kt & 1;
#if HAVE_TDM
        if (wid == 0) {
            if (kt + 1 < nk) {
                // DMA next tile into the other buffer; overlaps this tile's WMMAs.
                tdm_load_tile((unsigned)(unsigned long long)&bstage[cur ^ 1][0],
                              btile0 + (k0 + 32), 64, ldb);
                __builtin_amdgcn_s_wait_tensorcnt(1);   // current tile landed
            } else {
                __builtin_amdgcn_s_wait_tensorcnt(0);
            }
        }
        __syncthreads();                                // publish bstage[cur]
#else
        __syncthreads();
        for (int r = threadIdx.x >> 2; r < 64; r += 32) {
            int ch = threadIdx.x & 3;
            *reinterpret_cast<uint4*>(&bstage[0][r * 64 + ch * 16]) =
                *reinterpret_cast<const uint4*>(btile0 + (size_t)r * ldb + k0 + ch * 8);
        }
        __syncthreads();
        cur = 0;
#endif
        v16bf af = load_frag(arow, k0 + kb);
        __builtin_prefetch(arow + k0 + 32, 0, 1);
        const __bf16* lb = (const __bf16*)&bstage[cur][0];  // row stride 32 elems
#pragma unroll
        for (int j = 0; j < 4; ++j) {
            v16bf bfrag = load_frag(lb + (j * 16 + ml) * 32, kb);
            acc[j] = __builtin_amdgcn_wmma_f32_16x16x32_bf16(
                false, af, false, bfrag, (short)0, acc[j], false, false);
        }
#if HAVE_TDM
        __syncthreads();   // all waves done reading bstage[cur] before reuse
#endif
    }

#pragma unroll
    for (int j = 0; j < 4; ++j) {
#pragma unroll
        for (int r = 0; r < 8; ++r) {
            int row = tm + r + ((lane >> 4) * 8);
            int col = tn + j * 16 + ml;
            float v = acc[j][r];
            if (EPI & EPI_BIAS) v += bias[col];
            if (EPI & EPI_GATE) v *= gatep[(size_t)row * ldaux + col];
            if (EPI & EPI_RES)  v += resp[(size_t)row * ldaux + col];
            if (EPI & EPI_RELU) v = fmaxf(v, 0.f);
            if (EPI & EPI_SIG)  v = 1.f / (1.f + expf(-v));
            if (EPI & EPI_WF32)  Cf[coff + (size_t)row * ldcf + col] = v;
            if (EPI & EPI_WBF16) Cbf[coff + (size_t)row * ldcbf + col] = f2bf(v);
        }
    }
}

// -------------------- LayerNorm (row of length D) ---------------------------
__global__ void ln_k(const float* __restrict__ x, const float* __restrict__ g,
                     const float* __restrict__ b, float* __restrict__ outf,
                     __bf16* __restrict__ outbf, int D)
{
    size_t base = (size_t)blockIdx.x * D;
    const float* xr = x + base;
    float s = 0.f, s2 = 0.f;
    for (int c = threadIdx.x; c < D; c += blockDim.x) {
        float v = xr[c];
        s += v; s2 += v * v;
    }
    s  = blockReduceSum(s);
    s2 = blockReduceSum(s2);
    float mean = s / D;
    float var  = s2 / D - mean * mean;
    float rstd = rsqrtf(var + 1e-5f);
    for (int c = threadIdx.x; c < D; c += blockDim.x) {
        float y = (xr[c] - mean) * rstd * g[c] + b[c];
        if (outf)  outf[base + c] = y;
        if (outbf) outbf[base + c] = f2bf(y);
    }
}

// -------------------- delta / B_t projections (Ds = 16 each) ----------------
__global__ void delta_bt_k(const float* __restrict__ xn,
                           const float* __restrict__ Wd, const float* __restrict__ bd,
                           const float* __restrict__ Wi, const float* __restrict__ bi,
                           float* __restrict__ delta, float* __restrict__ Btout,
                           int D, int Ds)
{
    __shared__ float rowbuf[1024];
    size_t base = (size_t)blockIdx.x * D;
    for (int c = threadIdx.x; c < D; c += blockDim.x) rowbuf[c] = xn[base + c];
    __syncthreads();

    int lane = threadIdx.x & 31;
    int wave = threadIdx.x >> 5;             // 8 waves, 4 outputs each => 32
    for (int oi = 0; oi < 4; ++oi) {
        int o = wave * 4 + oi;               // 0..31
        const float* W = (o < Ds) ? Wd : Wi;
        int col = (o < Ds) ? o : (o - Ds);
        float sum = 0.f;
        for (int i = lane; i < D; i += 32) sum += rowbuf[i] * W[(size_t)i * Ds + col];
        sum = waveReduceSum(sum);
        if (lane == 0) {
            if (o < Ds) delta[(size_t)blockIdx.x * Ds + col] = 1.f / (1.f + expf(-(sum + bd[col])));
            else        Btout[(size_t)blockIdx.x * Ds + col] = sum + bi[col];
        }
    }
}

// -------------------- sequential SSM scan (one wave per batch) --------------
// h_t = tanh((h_{t-1} * d_t) @ A + B_t), Ds = 16, state in lanes 0..15.
__global__ void scan_k(const float* __restrict__ delta, const float* __restrict__ Bt,
                       const float* __restrict__ Amat, __bf16* __restrict__ hpad,
                       int S, int Ds)
{
    int lane = threadIdx.x & 31;
    int b = threadIdx.x >> 5;                // 4 waves = 4 batches
    int j = lane & 15;
    float Areg[16];
#pragma unroll
    for (int i = 0; i < 16; ++i) Areg[i] = Amat[i * 16 + j];

    float h = 0.f;
    for (int t = 0; t < S; ++t) {
        size_t idx = ((size_t)b * S + t) * Ds + j;
        float d  = delta[idx];
        float bt = Bt[idx];
        float p = h * d;                     // all 32 lanes run uniformly
        float acc = bt;
#pragma unroll
        for (int i = 0; i < 16; ++i) acc += __shfl(p, i, 32) * Areg[i];
        h = tanhf(acc);
        size_t obase = ((size_t)b * S + t) * 32;
        if (lane < 16) hpad[obase + j] = f2bf(h);        // real state
        else           hpad[obase + lane] = f2bf(0.f);   // K-padding to 32
    }
}

// -------------------- fp32 -> bf16 weight transpose (K,N)->(N,Kpad) ---------
__global__ void transpose_w(const float* __restrict__ in, __bf16* __restrict__ out,
                            int K, int N, int Kpad)
{
    size_t idx = (size_t)blockIdx.x * blockDim.x + threadIdx.x;
    size_t total = (size_t)N * Kpad;
    if (idx >= total) return;
    int n = (int)(idx / Kpad);
    int k = (int)(idx % Kpad);
    float v = (k < K) ? in[(size_t)k * N + n] : 0.f;
    out[idx] = f2bf(v);
}

// -------------------- per-head V transpose: (B,S,H,hd) -> (B,H,hd,S) --------
__global__ void vt_k(const __bf16* __restrict__ in, __bf16* __restrict__ out,
                     int Bn, int S, int D, int H, int hd)
{
    size_t idx = (size_t)blockIdx.x * blockDim.x + threadIdx.x;
    size_t total = (size_t)Bn * S * D;
    if (idx >= total) return;
    int d = (int)(idx % D);
    size_t t = idx / D;
    int s = (int)(t % S);
    int b = (int)(t / S);
    int h = d / hd, dd = d % hd;
    out[(((size_t)b * H + h) * hd + dd) * S + s] = in[idx];
}

// -------------------- softmax over rows of length L (=2048) -----------------
__global__ void softmax_k(const float* __restrict__ sc, __bf16* __restrict__ attn,
                          int L, float scale)
{
    size_t base = (size_t)blockIdx.x * L;
    int t = threadIdx.x;                      // 256 threads, 8 elems each
    float vals[8];
    float vmax = -3.0e38f;
#pragma unroll
    for (int i = 0; i < 8; ++i) {
        vals[i] = sc[base + t + i * 256] * scale;
        vmax = fmaxf(vmax, vals[i]);
    }
    vmax = blockReduceMax(vmax);
    float s = 0.f;
#pragma unroll
    for (int i = 0; i < 8; ++i) { vals[i] = expf(vals[i] - vmax); s += vals[i]; }
    s = blockReduceSum(s);
    float inv = 1.f / s;
#pragma unroll
    for (int i = 0; i < 8; ++i) attn[base + t + i * 256] = f2bf(vals[i] * inv);
}

// ---------------------------------------------------------------------------
extern "C" void kernel_launch(void* const* d_in, const int* in_sizes, int n_in,
                              void* d_out, int out_size, void* d_ws, size_t ws_size,
                              hipStream_t stream)
{
    (void)in_sizes; (void)n_in; (void)out_size; (void)ws_size;
    const int Bn = 4, S = 2048, D = 1024, Ds = 16, DFF = 4096, H = 8, hd = 128;
    const int M = Bn * S;                 // 8192 rows

    const float* x      = (const float*)d_in[0];
    const float* ln_g   = (const float*)d_in[1];
    const float* ln_b   = (const float*)d_in[2];
    const float* Wdelta = (const float*)d_in[3];
    const float* bdelta = (const float*)d_in[4];
    const float* Win    = (const float*)d_in[5];
    const float* b_in   = (const float*)d_in[6];
    const float* Wgate  = (const float*)d_in[7];
    const float* bgate  = (const float*)d_in[8];
    const float* Amat   = (const float*)d_in[9];
    const float* Cmat   = (const float*)d_in[10];
    const float* ln1_g  = (const float*)d_in[11];
    const float* ln1_b  = (const float*)d_in[12];
    const float* Wq     = (const float*)d_in[13];
    const float* bq     = (const float*)d_in[14];
    const float* Wk     = (const float*)d_in[15];
    const float* bk     = (const float*)d_in[16];
    const float* Wv     = (const float*)d_in[17];
    const float* bv     = (const float*)d_in[18];
    const float* Wo     = (const float*)d_in[19];
    const float* bo     = (const float*)d_in[20];
    const float* ln2_g  = (const float*)d_in[21];
    const float* ln2_b  = (const float*)d_in[22];
    const float* W1     = (const float*)d_in[23];
    const float* b1     = (const float*)d_in[24];
    const float* W2     = (const float*)d_in[25];
    const float* b2     = (const float*)d_in[26];
    const float* Wf     = (const float*)d_in[27];
    const float* bF     = (const float*)d_in[28];
    const float* lnf_g  = (const float*)d_in[29];
    const float* lnf_b  = (const float*)d_in[30];

    // ---- workspace carve (256B aligned) ----
    size_t off = 0;
    auto alloc = [&](size_t bytes) -> void* {
        void* p = (void*)((char*)d_ws + off);
        off += (bytes + 255) & ~(size_t)255;
        return p;
    };
    float*  xn_f    = (float*) alloc((size_t)M * D * 4);
    __bf16* xn_bf   = (__bf16*)alloc((size_t)M * D * 2);
    __bf16* xn1_bf  = (__bf16*)alloc((size_t)M * D * 2);
    float*  delta   = (float*) alloc((size_t)M * Ds * 4);
    float*  Btb     = (float*) alloc((size_t)M * Ds * 4);
    float*  gate    = (float*) alloc((size_t)M * D * 4);
    __bf16* hpad    = (__bf16*)alloc((size_t)M * 32 * 2);
    __bf16* combined= (__bf16*)alloc((size_t)M * 2 * D * 2);
    __bf16* qbf     = (__bf16*)alloc((size_t)M * D * 2);
    __bf16* kbf     = (__bf16*)alloc((size_t)M * D * 2);
    __bf16* vbf     = (__bf16*)alloc((size_t)M * D * 2);
    __bf16* vtb     = (__bf16*)alloc((size_t)M * D * 2);
    float*  scores  = (float*) alloc((size_t)Bn * H * S * S * 4);
    __bf16* attn    = (__bf16*)alloc((size_t)Bn * H * S * S * 2);
    __bf16* ctxbf   = (__bf16*)alloc((size_t)M * D * 2);
    float*  x2      = (float*) alloc((size_t)M * D * 4);
    __bf16* xn2_bf  = (__bf16*)alloc((size_t)M * D * 2);
    __bf16* hff     = (__bf16*)alloc((size_t)M * DFF * 2);
    float*  fbuf    = (float*) alloc((size_t)M * D * 4);
    __bf16* WtGate  = (__bf16*)alloc((size_t)D * D * 2);
    __bf16* WtQ     = (__bf16*)alloc((size_t)D * D * 2);
    __bf16* WtK     = (__bf16*)alloc((size_t)D * D * 2);
    __bf16* WtV     = (__bf16*)alloc((size_t)D * D * 2);
    __bf16* WtO     = (__bf16*)alloc((size_t)D * D * 2);
    __bf16* Wt1     = (__bf16*)alloc((size_t)DFF * D * 2);
    __bf16* Wt2     = (__bf16*)alloc((size_t)D * DFF * 2);
    __bf16* WtF     = (__bf16*)alloc((size_t)D * 2 * D * 2);
    __bf16* CtPad   = (__bf16*)alloc((size_t)D * 32 * 2);

    auto tgrid = [](size_t total) { return dim3((unsigned)((total + 255) / 256)); };

    // ---- weight prep: (K,N) fp32 -> (N,Kpad) bf16 ----
    transpose_w<<<tgrid((size_t)D * D), 256, 0, stream>>>(Wgate, WtGate, D, D, D);
    transpose_w<<<tgrid((size_t)D * D), 256, 0, stream>>>(Wq, WtQ, D, D, D);
    transpose_w<<<tgrid((size_t)D * D), 256, 0, stream>>>(Wk, WtK, D, D, D);
    transpose_w<<<tgrid((size_t)D * D), 256, 0, stream>>>(Wv, WtV, D, D, D);
    transpose_w<<<tgrid((size_t)D * D), 256, 0, stream>>>(Wo, WtO, D, D, D);
    transpose_w<<<tgrid((size_t)DFF * D), 256, 0, stream>>>(W1, Wt1, D, DFF, D);
    transpose_w<<<tgrid((size_t)D * DFF), 256, 0, stream>>>(W2, Wt2, DFF, D, DFF);
    transpose_w<<<tgrid((size_t)D * 2 * D), 256, 0, stream>>>(Wf, WtF, 2 * D, D, 2 * D);
    transpose_w<<<tgrid((size_t)D * 32), 256, 0, stream>>>(Cmat, CtPad, Ds, D, 32);

    // ---- mamba branch ----
    ln_k<<<M, 256, 0, stream>>>(x, ln_g, ln_b, xn_f, xn_bf, D);
    delta_bt_k<<<M, 256, 0, stream>>>(xn_f, Wdelta, bdelta, Win, b_in, delta, Btb, D, Ds);
    gemm_k<EPI_BIAS | EPI_SIG | EPI_WF32><<<dim3(M / 64, D / 64, 1), 128, 0, stream>>>(
        xn_bf, WtGate, bgate, nullptr, nullptr, 0, gate, D, nullptr, 0,
        M, D, D, D, D, 1, 0, 0, 0, 0, 0, 0);
    scan_k<<<1, 128, 0, stream>>>(delta, Btb, Amat, hpad, S, Ds);
    // mamba_out = (h @ C) * gate + x  -> bf16 into combined[:, :D]
    gemm_k<EPI_GATE | EPI_RES | EPI_WBF16><<<dim3(M / 64, D / 64, 1), 128, 0, stream>>>(
        hpad, CtPad, nullptr, gate, x, D, nullptr, 0, combined, 2 * D,
        M, D, 32, 32, 32, 1, 0, 0, 0, 0, 0, 0);

    // ---- transformer branch ----
    ln_k<<<M, 256, 0, stream>>>(x, ln1_g, ln1_b, nullptr, xn1_bf, D);
    gemm_k<EPI_BIAS | EPI_WBF16><<<dim3(M / 64, D / 64, 1), 128, 0, stream>>>(
        xn1_bf, WtQ, bq, nullptr, nullptr, 0, nullptr, 0, qbf, D,
        M, D, D, D, D, 1, 0, 0, 0, 0, 0, 0);
    gemm_k<EPI_BIAS | EPI_WBF16><<<dim3(M / 64, D / 64, 1), 128, 0, stream>>>(
        xn1_bf, WtK, bk, nullptr, nullptr, 0, nullptr, 0, kbf, D,
        M, D, D, D, D, 1, 0, 0, 0, 0, 0, 0);
    gemm_k<EPI_BIAS | EPI_WBF16><<<dim3(M / 64, D / 64, 1), 128, 0, stream>>>(
        xn1_bf, WtV, bv, nullptr, nullptr, 0, nullptr, 0, vbf, D,
        M, D, D, D, D, 1, 0, 0, 0, 0, 0, 0);
    vt_k<<<tgrid((size_t)M * D), 256, 0, stream>>>(vbf, vtb, Bn, S, D, H, hd);

    // scores[b,h] = Q_head @ K_head^T   (batched over 32 (b,h) pairs)
    gemm_k<EPI_WF32><<<dim3(S / 64, S / 64, Bn * H), 128, 0, stream>>>(
        qbf, kbf, nullptr, nullptr, nullptr, 0, scores, S, nullptr, 0,
        S, S, hd, D, D,
        H, (long)S * D, (long)hd, (long)S * D, (long)hd,
        (long)H * S * S, (long)S * S);
    softmax_k<<<Bn * H * S, 256, 0, stream>>>(scores, attn, S, 0.0883883476483f);
    // ctx[b,h] = attn @ V_head  (B operand = V^T per head)
    gemm_k<EPI_WBF16><<<dim3(S / 64, hd / 64, Bn * H), 128, 0, stream>>>(
        attn, vtb, nullptr, nullptr, nullptr, 0, nullptr, 0, ctxbf, D,
        S, hd, S, S, S,
        H, (long)H * S * S, (long)S * S, (long)H * hd * S, (long)hd * S,
        (long)S * D, (long)hd);
    // x2 = x + ctx @ Wo + bo
    gemm_k<EPI_BIAS | EPI_RES | EPI_WF32><<<dim3(M / 64, D / 64, 1), 128, 0, stream>>>(
        ctxbf, WtO, bo, nullptr, x, D, x2, D, nullptr, 0,
        M, D, D, D, D, 1, 0, 0, 0, 0, 0, 0);
    ln_k<<<M, 256, 0, stream>>>(x2, ln2_g, ln2_b, nullptr, xn2_bf, D);
    gemm_k<EPI_BIAS | EPI_RELU | EPI_WBF16><<<dim3(M / 64, DFF / 64, 1), 128, 0, stream>>>(
        xn2_bf, Wt1, b1, nullptr, nullptr, 0, nullptr, 0, hff, DFF,
        M, DFF, D, D, D, 1, 0, 0, 0, 0, 0, 0);
    // t_out = x2 + hff @ W2 + b2 -> bf16 into combined[:, D:]
    gemm_k<EPI_BIAS | EPI_RES | EPI_WBF16><<<dim3(M / 64, D / 64, 1), 128, 0, stream>>>(
        hff, Wt2, b2, nullptr, x2, D, nullptr, 0, combined + D, 2 * D,
        M, D, DFF, DFF, DFF, 1, 0, 0, 0, 0, 0, 0);

    // ---- fusion ----
    gemm_k<EPI_BIAS | EPI_WF32><<<dim3(M / 64, D / 64, 1), 128, 0, stream>>>(
        combined, WtF, bF, nullptr, nullptr, 0, fbuf, D, nullptr, 0,
        M, D, 2 * D, 2 * D, 2 * D, 1, 0, 0, 0, 0, 0, 0);
    ln_k<<<M, 256, 0, stream>>>(fbuf, lnf_g, lnf_b, (float*)d_out, nullptr, D);
}
